// ArbSR_76665166233860
// MI455X (gfx1250) — matrix operand order repaired and checked by reading, not compiled, not run on hardware
//
#include <hip/hip_runtime.h>
#include <hip/hip_bf16.h>
#include <math.h>

// ---------------------------------------------------------------------------
// ArbSR pipeline for gfx1250 (MI455X), wave32 / WMMA.
//   feat grid 128x128, SCALE=4 -> 512x512 output grid, C=64, E=4, Q=262144.
//
// Key observation: with SCALE=4 the body-MLP inputs (cs, cs, coor_h, coor_w)
// are exactly periodic with period 4 in each spatial dim (all fp ops involved
// are exact), so emb2 / routing r / offset off take only 16 distinct values,
// one per phase (hh%4, ww%4). The 64x64 GEMM is computed once as a single
// WMMA tile (N = 16 phases) by one wave; the per-pixel kernel reads r/off
// from a 96-float LDS table. This removes the register-spill-bound tile loop
// of the previous round (no loop -> no LICM pressure around the WMMAs).
// ---------------------------------------------------------------------------

#define HF 128
#define WF 128
#define FEAT_HW (HF * WF)      // 16384
#define HO 512
#define WO 512
#define HWO (HO * WO)          // 262144

typedef __attribute__((ext_vector_type(16))) _Float16 v16h;
typedef __attribute__((ext_vector_type(8)))  float    v8f;

__device__ __forceinline__ float sigmoidf_(float x) {
    return 1.0f / (1.0f + __expf(-x));
}

// --------------------------------------------------------------------- enc
__global__ void __launch_bounds__(256, 1)
enc_conv_kernel(const float* __restrict__ inp,
                const float* __restrict__ ew,
                const float* __restrict__ eb,
                float* __restrict__ feat) {
    __shared__ __align__(16) float sW[64 * 27];
    __shared__ float sB[64];
    int tid = threadIdx.x;
    for (int i = tid; i < 64 * 27; i += blockDim.x) sW[i] = ew[i];
    for (int i = tid; i < 64; i += blockDim.x) sB[i] = eb[i];
    __syncthreads();

    int p = blockIdx.x * blockDim.x + tid;
    if (p >= FEAT_HW) return;
    int y = p >> 7, x = p & 127;

    float patch[27];
#pragma unroll
    for (int ci = 0; ci < 3; ++ci)
#pragma unroll
        for (int dy = 0; dy < 3; ++dy)
#pragma unroll
            for (int dx = 0; dx < 3; ++dx) {
                int yy = y + dy - 1, xx = x + dx - 1;
                bool ok = (yy >= 0) && (yy < HF) && (xx >= 0) && (xx < WF);
                patch[ci * 9 + dy * 3 + dx] =
                    ok ? inp[ci * FEAT_HW + yy * WF + xx] : 0.0f;
            }

    for (int o = 0; o < 64; ++o) {
        float s = sB[o];
#pragma unroll
        for (int j = 0; j < 27; ++j) s += sW[o * 27 + j] * patch[j];
        feat[o * FEAT_HW + p] = s;
    }
}

// ------------------------------------------------------------ body (WMMA)
// One wave, one WMMA N-tile: columns = 16 phases (py*4+px).
// emb1[c,phase] = relu(b1[c] + (w1[c,0]+w1[c,1])*0.25 + w1[c,2]*ch + w1[c,3]*cw)
// emb2 = relu(W2 @ emb1 + b2)    -> v_wmma_f32_16x16x32_f16 (4 M-tiles x 2 K)
// rtab = sigmoid(RW @ emb2 + rb) (4x16)   otab = OW @ emb2 + ob (2x16)
__global__ void body_wmma_kernel(const float* __restrict__ w1,
                                 const float* __restrict__ b1,
                                 const float* __restrict__ w2,
                                 const float* __restrict__ b2,
                                 const float* __restrict__ rw,
                                 const float* __restrict__ rb,
                                 const float* __restrict__ ow,
                                 const float* __restrict__ ob,
                                 float* __restrict__ rtab,   // [4][16]
                                 float* __restrict__ otab) { // [2][16]
    int lane = threadIdx.x & 31;
    int lo = lane & 15;
    int hi = (lane >> 4) & 1;

    // A fragments: W2[64][64] as 4 M-tiles x 2 K-chunks (f16).
    // A element e -> M = 16t + (lane&15), K = 32k + 8*hi + (e<8 ? e : e+8)
    v16h afrag[4][2];
#pragma unroll
    for (int t = 0; t < 4; ++t) {
        int m = 16 * t + lo;
#pragma unroll
        for (int k = 0; k < 2; ++k) {
#pragma unroll
            for (int e = 0; e < 16; ++e) {
                int c = k * 32 + 8 * hi + (e < 8 ? e : e + 8);
                afrag[t][k][e] = (_Float16)w2[m * 64 + c];
            }
        }
    }

    // Phase coords for this lane's column: py = lo>>2, px = lo&3.
    // coor(i) = (i+0.5)/4 - floor((i+0.5)/4 + 0.001) - 0.5 == (p+0.5)*0.25-0.5
    float ch = ((lo >> 2) + 0.5f) * 0.25f - 0.5f;
    float cw = ((lo & 3) + 0.5f) * 0.25f - 0.5f;

    // B fragments (emb1): element e -> N = lane&15, K = 32k + 8*hi + (e<8 ? e : e+8)
    v16h bfrag[2];
#pragma unroll
    for (int k = 0; k < 2; ++k) {
#pragma unroll
        for (int e = 0; e < 16; ++e) {
            int c = k * 32 + hi * 8 + (e < 8 ? e : e + 8);
            float v = b1[c] + (w1[c * 4 + 0] + w1[c * 4 + 1]) * 0.25f
                    + w1[c * 4 + 2] * ch + w1[c * 4 + 3] * cw;
            bfrag[k][e] = (_Float16)fmaxf(v, 0.0f);
        }
    }

    // D element v -> M = 16t + v + 8*hi ; seed accumulator with bias b2.
    v8f acc[4];
#pragma unroll
    for (int t = 0; t < 4; ++t)
#pragma unroll
        for (int v = 0; v < 8; ++v) acc[t][v] = b2[16 * t + 8 * hi + v];

#pragma unroll
    for (int t = 0; t < 4; ++t) {
        acc[t] = __builtin_amdgcn_wmma_f32_16x16x32_f16(
            false, afrag[t][0], false, bfrag[0], (short)0, acc[t], false, false);
        acc[t] = __builtin_amdgcn_wmma_f32_16x16x32_f16(
            false, afrag[t][1], false, bfrag[1], (short)0, acc[t], false, false);
    }

#pragma unroll
    for (int t = 0; t < 4; ++t)
#pragma unroll
        for (int v = 0; v < 8; ++v) acc[t][v] = fmaxf(acc[t][v], 0.0f);

    // Heads: 4 routing + 2 offset dots over this lane's 32 channels
    // (M = 16t + 8*hi + v), completed across half-waves with shfl_xor(16).
    float p[6];
#pragma unroll
    for (int j = 0; j < 6; ++j) p[j] = 0.0f;
#pragma unroll
    for (int t = 0; t < 4; ++t) {
        int mb = 16 * t + 8 * hi;
#pragma unroll
        for (int j = 0; j < 6; ++j) {
            const float* Wj = (j < 4) ? (rw + j * 64) : (ow + (j - 4) * 64);
#pragma unroll
            for (int v = 0; v < 8; ++v) p[j] += Wj[mb + v] * acc[t][v];
        }
    }
#pragma unroll
    for (int j = 0; j < 6; ++j) p[j] += __shfl_xor(p[j], 16);

    if (lane < 16) {
#pragma unroll
        for (int e = 0; e < 4; ++e)
            rtab[e * 16 + lo] = sigmoidf_(p[e] + rb[e]);
        otab[lo]      = p[4] + ob[0];
        otab[16 + lo] = p[5] + ob[1];
    }
}

// ------------------------------------------------------------------- fuse
// bilinear gather of feat (zero-pad) + rank-8 expert mixture, z = out2+fea0.
__global__ void __launch_bounds__(256, 1)
fuse_kernel(const float* __restrict__ feat,
            const float* __restrict__ rtab,  // [4][16]
            const float* __restrict__ otab,  // [2][16]
            const float* __restrict__ wc,    // [4][8][64]
            const float* __restrict__ we,    // [4][64][8]
            float* __restrict__ z) {
    __shared__ __align__(16) float sWc[2048];
    __shared__ __align__(16) float sWe[2048];
    __shared__ float sR[64];
    __shared__ float sO[32];
    int tid = threadIdx.x;
    for (int i = tid; i < 2048; i += blockDim.x) { sWc[i] = wc[i]; sWe[i] = we[i]; }
    if (tid < 64) sR[tid] = rtab[tid];
    if (tid < 32) sO[tid] = otab[tid];
    __syncthreads();

    int n = blockIdx.x * blockDim.x + tid;  // grid sized exactly HWO
    int hh = n >> 9, ww = n & 511;
    int phase = ((hh & 3) << 2) | (ww & 3);

    float o0 = sO[phase], o1 = sO[16 + phase];
    const float inv = 2.0f / 127.0f;
    float gx = ((ww + 0.5f) * 0.25f - 0.5f) * inv - 1.0f + o0 * inv;
    float gy = ((hh + 0.5f) * 0.25f - 0.5f) * inv - 1.0f + o1 * inv;
    float x = (gx + 1.0f) * 0.5f * 127.0f;
    float y = (gy + 1.0f) * 0.5f * 127.0f;
    float x0f = floorf(x), y0f = floorf(y);
    float wx1 = x - x0f, wx0 = 1.0f - wx1;
    float wy1 = y - y0f, wy0 = 1.0f - wy1;
    int x0 = (int)x0f, y0 = (int)y0f;
    int xs[2] = {x0, x0 + 1}, ys[2] = {y0, y0 + 1};
    float wxs[2] = {wx0, wx1}, wys[2] = {wy0, wy1};

    float cwt[4];
    int cb[4];
#pragma unroll
    for (int j = 0; j < 2; ++j)
#pragma unroll
        for (int i = 0; i < 2; ++i) {
            int yi = ys[j], xi = xs[i];
            bool ok = (xi >= 0) && (xi < WF) && (yi >= 0) && (yi < HF);
            int yc = min(max(yi, 0), HF - 1), xc = min(max(xi, 0), WF - 1);
            cwt[j * 2 + i] = ok ? (wys[j] * wxs[i]) : 0.0f;
            cb[j * 2 + i] = yc * WF + xc;
        }

    float fea[64];
#pragma unroll
    for (int c = 0; c < 64; ++c) {
        const float* fp = feat + c * FEAT_HW;
        fea[c] = cwt[0] * fp[cb[0]] + cwt[1] * fp[cb[1]]
               + cwt[2] * fp[cb[2]] + cwt[3] * fp[cb[3]];
    }

    float re[4];
#pragma unroll
    for (int e = 0; e < 4; ++e) re[e] = sR[e * 16 + phase];

    float mid[8];
#pragma unroll
    for (int k = 0; k < 8; ++k) mid[k] = 0.0f;

    for (int e = 0; e < 4; ++e) {
#pragma unroll
        for (int k = 0; k < 8; ++k) {
            const float4* w4 = reinterpret_cast<const float4*>(sWc + (e * 8 + k) * 64);
            float dot = 0.0f;
#pragma unroll
            for (int c4 = 0; c4 < 16; ++c4) {
                float4 wv = w4[c4];
                dot += wv.x * fea[c4 * 4 + 0] + wv.y * fea[c4 * 4 + 1]
                     + wv.z * fea[c4 * 4 + 2] + wv.w * fea[c4 * 4 + 3];
            }
            mid[k] += re[e] * dot;
        }
    }

    for (int e = 0; e < 4; ++e) {
        float rv = re[e];
#pragma unroll
        for (int c = 0; c < 64; ++c) {
            const float4* w4 = reinterpret_cast<const float4*>(sWe + (e * 64 + c) * 8);
            float4 wa = w4[0], wb = w4[1];
            float d = wa.x * mid[0] + wa.y * mid[1] + wa.z * mid[2] + wa.w * mid[3]
                    + wb.x * mid[4] + wb.y * mid[5] + wb.z * mid[6] + wb.w * mid[7];
            fea[c] += rv * d;  // z = out2 + fea0, accumulated in place
        }
    }

#pragma unroll
    for (int c = 0; c < 64; ++c) z[c * HWO + n] = fea[c];
}

// -------------------------------------------------------------- tail conv
__global__ void __launch_bounds__(256, 1)
tail_conv_kernel(const float* __restrict__ z,
                 const float* __restrict__ tw,  // [3][64][3][3]
                 const float* __restrict__ tb,
                 float* __restrict__ pred) {
    __shared__ __align__(16) float sWT[9 * 64 * 3];  // [k][c][o]
    int tid = threadIdx.x;
    for (int i = tid; i < 1728; i += blockDim.x) {
        int o = i / 576, rem = i % 576, c = rem / 9, k = rem % 9;
        sWT[k * 192 + c * 3 + o] = tw[i];
    }
    __syncthreads();

    int n = blockIdx.x * blockDim.x + tid;
    int y = n >> 9, x = n & 511;
    float a0 = tb[0], a1 = tb[1], a2 = tb[2];
    for (int ky = 0; ky < 3; ++ky) {
        int yy = y + ky - 1;
        if (yy < 0 || yy >= HO) continue;
        for (int kx = 0; kx < 3; ++kx) {
            int xx = x + kx - 1;
            if (xx < 0 || xx >= WO) continue;
            const float* wp = sWT + (ky * 3 + kx) * 192;
            const float* zp = z + yy * WO + xx;
#pragma unroll
            for (int c = 0; c < 64; ++c) {
                float v = zp[c * HWO];
                a0 += v * wp[c * 3 + 0];
                a1 += v * wp[c * 3 + 1];
                a2 += v * wp[c * 3 + 2];
            }
        }
    }
    pred[n] = a0;
    pred[HWO + n] = a1;
    pred[2 * HWO + n] = a2;
}

// ----------------------------------------------------------------- gather
__global__ void gather_kernel(const float* __restrict__ coord,
                              const float* __restrict__ cell,
                              const float* __restrict__ pred,
                              float* __restrict__ out) {
    int q = blockIdx.x * blockDim.x + threadIdx.x;  // Q == HWO exactly
    float cy = coord[q * 2 + 0], cx = coord[q * 2 + 1];
    float c0 = cell[q * 2 + 0], c1 = cell[q * 2 + 1];
    const float lo = -1.0f + 1e-6f, hi = 1.0f - 1e-6f;
    float gyq = fminf(fmaxf(cy - c0 * 0.5f + 1e-6f, lo), hi);
    float gxq = fminf(fmaxf(cx - c1 * 0.5f + 1e-6f, lo), hi);
    int xi = (int)roundf((gxq + 1.0f) * 0.5f * (float)(WO - 1));
    int yi = (int)roundf((gyq + 1.0f) * 0.5f * (float)(HO - 1));
    xi = min(max(xi, 0), WO - 1);
    yi = min(max(yi, 0), HO - 1);
    int base = yi * WO + xi;
    out[q * 3 + 0] = pred[base];
    out[q * 3 + 1] = pred[HWO + base];
    out[q * 3 + 2] = pred[2 * HWO + base];
}

// ------------------------------------------------------------------ launch
extern "C" void kernel_launch(void* const* d_in, const int* in_sizes, int n_in,
                              void* d_out, int out_size, void* d_ws, size_t ws_size,
                              hipStream_t stream) {
    const float* inp   = (const float*)d_in[0];
    const float* coord = (const float*)d_in[1];
    const float* cell  = (const float*)d_in[2];
    const float* enc_w = (const float*)d_in[3];
    const float* enc_b = (const float*)d_in[4];
    const float* w1    = (const float*)d_in[5];
    const float* b1    = (const float*)d_in[6];
    const float* w2    = (const float*)d_in[7];
    const float* b2    = (const float*)d_in[8];
    const float* rw    = (const float*)d_in[9];
    const float* rb    = (const float*)d_in[10];
    const float* ow    = (const float*)d_in[11];
    const float* ob    = (const float*)d_in[12];
    const float* tw    = (const float*)d_in[13];
    const float* tb    = (const float*)d_in[14];
    const float* wc    = (const float*)d_in[15];
    const float* we    = (const float*)d_in[16];
    float* out = (float*)d_out;

    float* ws   = (float*)d_ws;
    float* feat = ws;                          // 64*16384 = 1,048,576 floats
    float* rtab = feat + (size_t)64 * FEAT_HW; // 64 floats
    float* otab = rtab + 64;                   // 32 floats
    float* z    = otab + 32;                   // 64*HWO = 16,777,216 floats
    float* pred = z + (size_t)64 * HWO;        // 3*HWO  = 786,432 floats
    (void)in_sizes; (void)n_in; (void)out_size; (void)ws_size;

    enc_conv_kernel<<<FEAT_HW / 256, 256, 0, stream>>>(inp, enc_w, enc_b, feat);
    body_wmma_kernel<<<1, 32, 0, stream>>>(w1, b1, w2, b2, rw, rb, ow, ob, rtab, otab);
    fuse_kernel<<<HWO / 256, 256, 0, stream>>>(feat, rtab, otab, wc, we, z);
    tail_conv_kernel<<<HWO / 256, 256, 0, stream>>>(z, tw, tb, pred);
    gather_kernel<<<HWO / 256, 256, 0, stream>>>(coord, cell, pred, out);
}
